// EMHA_71717363909291
// MI455X (gfx1250) — matrix operand order repaired and checked by reading, not compiled, not run on hardware
//
#include <hip/hip_runtime.h>
#include <hip/hip_bf16.h>

// ---------------------------------------------------------------------------
// EMHA for MI455X (gfx1250): bf16 WMMA pipeline with async-to-LDS staging.
//   x --gemm1--> XR(bf16) --gemm2--> QKV(bf16) --flash-attn--> O(bf16)
//     --gemm3--> out(f32)
// Matmuls: v_wmma_f32_16x16x32_bf16 (wave32, f32 accumulate).
// Tile staging: global_load_async_to_lds_b128 (+ s_wait_asynccnt), with
// double-buffered LDS in the GEMMs to overlap fetch with WMMA.
// ---------------------------------------------------------------------------

typedef __bf16 v4bf  __attribute__((ext_vector_type(4)));
typedef __bf16 v8bf  __attribute__((ext_vector_type(8)));
typedef __bf16 v16bf __attribute__((ext_vector_type(16)));
typedef float  v8f   __attribute__((ext_vector_type(8)));

#define USE_ASYNC_LDS 1

static __device__ __forceinline__ v16bf cat8(v8bf lo, v8bf hi) {
  return __builtin_shufflevector(lo, hi, 0,1,2,3,4,5,6,7,8,9,10,11,12,13,14,15);
}

// Flat LDS pointer -> wave-relative LDS byte offset (ISA 10.2: LDS aperture
// uses addr[31:0]).
static __device__ __forceinline__ unsigned lds_off(const void* p) {
  return (unsigned)(unsigned long long)(uintptr_t)p;
}

// 16B memory -> LDS async copy (ASYNCcnt-tracked; EXEC per-lane).
static __device__ __forceinline__ void async_copy_b128(const void* gsrc, void* ldst) {
#if USE_ASYNC_LDS
  asm volatile("global_load_async_to_lds_b128 %0, %1, off"
               :: "v"(lds_off(ldst)),
                  "v"((unsigned long long)(uintptr_t)gsrc)
               : "memory");
#else
  *(v8bf*)ldst = *(const v8bf*)gsrc;
#endif
}

static __device__ __forceinline__ void wait_async_all() {
#if USE_ASYNC_LDS
  asm volatile("s_wait_asynccnt 0x0" ::: "memory");
#endif
}

// Convert + transpose weights: W[K][N] f32  ->  Wt[N][K] bf16
__global__ __launch_bounds__(256) void wt_convert(
    const float* __restrict__ W, __bf16* __restrict__ Wt, int K, int N)
{
  int idx = blockIdx.x * 256 + threadIdx.x;
  if (idx >= N * K) return;
  int n = idx / K;
  int k = idx - n * K;
  Wt[idx] = (__bf16)W[(size_t)k * N + n];
}

// ---------------------------------------------------------------------------
// Tiled GEMM: C[M][N] = A[M][K] @ Bt[N][K]^T + bias
// Workgroup tile 128(M) x 128(N), K-step 32, double-buffered LDS.
// 8 waves in a 4(M) x 2(N) grid; each wave owns 32x64 = eight 16x16 f32
// accumulators -> 8 WMMAs per K-step against 6 LDS fragment loads.
// All dims are multiples of the tile sizes for this problem (no bounds checks).
// ---------------------------------------------------------------------------
template<bool A_F32, bool OUT_F32>
__global__ __launch_bounds__(256) void gemm_bf16_wmma(
    const void* __restrict__ Aptr, const __bf16* __restrict__ Bt,
    const float* __restrict__ bias, void* __restrict__ Cptr,
    int M, int N, int K)
{
  // padded row stride: 40 halfs = 80B (16B-aligned rows, spreads banks)
  __shared__ alignas(16) __bf16 As[2][128 * 40];
  __shared__ alignas(16) __bf16 Bs[2][128 * 40];

  const int tid   = threadIdx.x;
  const int lane  = tid & 31;
  const int wave  = tid >> 5;
  const int l15   = lane & 15;
  const int half  = lane >> 4;
  const int waveM = wave & 3;   // * 32 rows
  const int waveN = wave >> 2;  // * 64 cols
  const int bn    = blockIdx.x; // N / 128
  const int bm    = blockIdx.y; // M / 128

  v8f acc[2][4] = {};

  const float*  Af = (const float*)Aptr;
  const __bf16* Ab = (const __bf16*)Aptr;

  auto stageA = [&](int k0, int buf) {
    if (A_F32) {
      // fp32 -> bf16 conversion must go through VGPRs
#pragma unroll
      for (int i = 0; i < 4; ++i) {
        int c   = tid * 4 + i;          // 1024 float4 chunks (128x32 f32)
        int row = c >> 3;
        int cg  = c & 7;
        float4 f = *(const float4*)(Af + (size_t)(bm * 128 + row) * K + k0 + cg * 4);
        v4bf h;
        h[0] = (__bf16)f.x; h[1] = (__bf16)f.y;
        h[2] = (__bf16)f.z; h[3] = (__bf16)f.w;
        *(v4bf*)(&As[buf][row * 40 + cg * 4]) = h;
      }
    } else {
#pragma unroll
      for (int i = 0; i < 2; ++i) {
        int c   = tid * 2 + i;          // 512 16B chunks (128x32 bf16)
        int row = c >> 2;
        int cg  = c & 3;
        async_copy_b128(Ab + (size_t)(bm * 128 + row) * K + k0 + cg * 8,
                        &As[buf][row * 40 + cg * 8]);
      }
    }
  };
  auto stageB = [&](int k0, int buf) {
#pragma unroll
    for (int i = 0; i < 2; ++i) {
      int c   = tid * 2 + i;            // 512 16B chunks (128x32 bf16)
      int row = c >> 2;
      int cg  = c & 3;
      async_copy_b128(Bt + (size_t)(bn * 128 + row) * K + k0 + cg * 8,
                      &Bs[buf][row * 40 + cg * 8]);
    }
  };

  const int nk = K >> 5;
  stageA(0, 0);
  stageB(0, 0);

  for (int it = 0; it < nk; ++it) {
    const int cur = it & 1;
    wait_async_all();            // my async writes into 'cur' done
    __syncthreads();             // everyone's writes done; prior reads of cur^1 done
    if (it + 1 < nk) {           // prefetch next tile into the other buffer
      stageA((it + 1) * 32, cur ^ 1);
      stageB((it + 1) * 32, cur ^ 1);
    }

    // ---- fragments (ISA 7.12.2 layouts) ----
    v16bf a[2];
#pragma unroll
    for (int mt = 0; mt < 2; ++mt) {
      const __bf16* p = &As[cur][(waveM * 32 + mt * 16 + l15) * 40 + half * 8];
      a[mt] = cat8(*(const v8bf*)p, *(const v8bf*)(p + 16));
    }
#pragma unroll
    for (int nt = 0; nt < 4; ++nt) {
      const __bf16* p = &Bs[cur][(waveN * 64 + nt * 16 + l15) * 40 + half * 8];
      v16bf b = cat8(*(const v8bf*)p, *(const v8bf*)(p + 16));
#pragma unroll
      for (int mt = 0; mt < 2; ++mt)
        acc[mt][nt] = __builtin_amdgcn_wmma_f32_16x16x32_bf16(
            false, a[mt], false, b, (short)0, acc[mt][nt], false, false);
    }
  }

  // ---- epilogue: bias + store (C layout: VGPR r -> M = r + half*8, N = l15) ----
#pragma unroll
  for (int mt = 0; mt < 2; ++mt) {
    int gm0 = bm * 128 + waveM * 32 + mt * 16 + half * 8;
#pragma unroll
    for (int nt = 0; nt < 4; ++nt) {
      int gn = bn * 128 + waveN * 64 + nt * 16 + l15;
      float bv = bias[gn];
#pragma unroll
      for (int r = 0; r < 8; ++r) {
        float v = acc[mt][nt][r] + bv;
        if (OUT_F32) ((float*)Cptr)[(size_t)(gm0 + r) * N + gn] = v;
        else         ((__bf16*)Cptr)[(size_t)(gm0 + r) * N + gn] = (__bf16)v;
      }
    }
  }
}

// ---------------------------------------------------------------------------
// Flash attention over one 128-row q-chunk of one (b, h, split) block.
// Q tile held as register A-fragment; 64-key K/V tiles streamed via LDS
// (K async; V transposed through VGPRs). Online softmax in exp2 domain;
// P converted C-layout -> A-layout through per-wave LDS.
// ---------------------------------------------------------------------------
__global__ __launch_bounds__(256) void attn_wmma(
    const __bf16* __restrict__ qkv, __bf16* __restrict__ O)
{
  __shared__ alignas(16) __bf16 Qs[128 * 40];      // [q][d], stride 40
  __shared__ alignas(16) __bf16 Ks[64 * 40];       // [key][d], stride 40
  __shared__ alignas(16) __bf16 Vt[32 * 72];       // [d][key], stride 72
  __shared__ alignas(16) __bf16 Ps[8 * 16 * 72];   // per-wave [q16][key64], stride 72

  const int tid  = threadIdx.x;
  const int lane = tid & 31;
  const int wave = tid >> 5;
  const int l15  = lane & 15;
  const int half = lane >> 4;

  const int bid = blockIdx.x;          // b*384 + h*32 + s*8 + qc
  const int qc  = bid & 7;
  const int s   = (bid >> 3) & 3;
  const int h   = (bid >> 5) % 12;
  const int b   = bid / 384;

  const size_t segTok = (size_t)b * 4096 + (size_t)s * 1024;

  // ---- stage Q tile: 128 x 32 (async) ----
#pragma unroll
  for (int i = 0; i < 2; ++i) {
    int c   = tid * 2 + i;
    int row = c >> 2;
    int cg  = c & 3;
    size_t tok = segTok + (size_t)qc * 128 + row;
    async_copy_b128(qkv + tok * 1152 + h * 32 + cg * 8,
                    &Qs[row * 40 + cg * 8]);
  }
  wait_async_all();
  __syncthreads();

  // Q A-fragment for this wave's 16 rows (registers for the whole loop)
  v16bf aq;
  {
    const __bf16* p = &Qs[(wave * 16 + l15) * 40 + half * 8];
    aq = cat8(*(const v8bf*)p, *(const v8bf*)(p + 16));
  }

  float mrow[8], lrow[8];
  v8f o0 = {}, o1 = {};
#pragma unroll
  for (int r = 0; r < 8; ++r) { mrow[r] = -3.0e38f; lrow[r] = 0.0f; }

  // 1/sqrt(hd) * log2(e): softmax carried in base-2 domain (v_exp_f32 path)
  const float cscale = 0.17677669529663687f * 1.4426950408889634f;

  __bf16* pwave = &Ps[wave * 16 * 72];

  for (int kt = 0; kt < 16; ++kt) {
    __syncthreads();
    // ---- stage K tile 64 x 32 (async; row-major: B-frag K-dim = d) ----
    {
      int row = tid >> 2;
      int cg  = tid & 3;
      size_t tok = segTok + (size_t)kt * 64 + row;
      async_copy_b128(qkv + tok * 1152 + 384 + h * 32 + cg * 8,
                      &Ks[row * 40 + cg * 8]);
    }
    // ---- stage V tile transposed: Vt[d][key] (needs VGPR round-trip) ----
    {
      int key = tid >> 2;
      int dg  = tid & 3;
      size_t tok = segTok + (size_t)kt * 64 + key;
      v8bf vv = *(const v8bf*)(qkv + tok * 1152 + 768 + h * 32 + dg * 8);
#pragma unroll
      for (int j = 0; j < 8; ++j)
        Vt[(dg * 8 + j) * 72 + key] = vv[j];
    }
    wait_async_all();
    __syncthreads();

    // ---- scores: S[16q x 64key] = Q @ K^T, four N-tiles ----
    v8f sc[4];
#pragma unroll
    for (int nt = 0; nt < 4; ++nt) {
      const __bf16* p = &Ks[(nt * 16 + l15) * 40 + half * 8];
      v16bf bk = cat8(*(const v8bf*)p, *(const v8bf*)(p + 16));
      v8f z = {};
      sc[nt] = __builtin_amdgcn_wmma_f32_16x16x32_bf16(
          false, aq, false, bk, (short)0, z, false, false);
    }

    // ---- online softmax (rows live in 16-lane halves; xor<=8 stays inside)
#pragma unroll
    for (int r = 0; r < 8; ++r) {
      float v0 = sc[0][r] * cscale, v1 = sc[1][r] * cscale;
      float v2 = sc[2][r] * cscale, v3 = sc[3][r] * cscale;
      float mx = fmaxf(fmaxf(v0, v1), fmaxf(v2, v3));
      mx = fmaxf(mx, __shfl_xor(mx, 1, 32));
      mx = fmaxf(mx, __shfl_xor(mx, 2, 32));
      mx = fmaxf(mx, __shfl_xor(mx, 4, 32));
      mx = fmaxf(mx, __shfl_xor(mx, 8, 32));
      float mnew  = fmaxf(mrow[r], mx);
      float alpha = exp2f(mrow[r] - mnew);
      mrow[r] = mnew;
      float p0 = exp2f(v0 - mnew), p1 = exp2f(v1 - mnew);
      float p2 = exp2f(v2 - mnew), p3 = exp2f(v3 - mnew);
      sc[0][r] = p0; sc[1][r] = p1; sc[2][r] = p2; sc[3][r] = p3;
      float rs = p0 + p1 + p2 + p3;
      rs += __shfl_xor(rs, 1, 32);
      rs += __shfl_xor(rs, 2, 32);
      rs += __shfl_xor(rs, 4, 32);
      rs += __shfl_xor(rs, 8, 32);
      lrow[r] = lrow[r] * alpha + rs;
      o0[r] *= alpha;
      o1[r] *= alpha;
    }

    // ---- P: C-layout -> per-wave LDS (bf16) -> A-layout fragments ----
#pragma unroll
    for (int nt = 0; nt < 4; ++nt)
#pragma unroll
      for (int r = 0; r < 8; ++r)
        pwave[(r + half * 8) * 72 + nt * 16 + l15] = (__bf16)sc[nt][r];
    asm volatile("s_wait_dscnt 0x0" ::: "memory");   // same-wave LDS RAW fence

    // ---- O += P @ V : two K=32 steps over the 64-key tile ----
#pragma unroll
    for (int kk = 0; kk < 2; ++kk) {
      const __bf16* pp = &pwave[l15 * 72 + kk * 32 + half * 8];
      v16bf pa = cat8(*(const v8bf*)pp, *(const v8bf*)(pp + 16));
      const __bf16* pv0 = &Vt[l15 * 72 + kk * 32 + half * 8];
      v16bf vb0 = cat8(*(const v8bf*)pv0, *(const v8bf*)(pv0 + 16));
      const __bf16* pv1 = &Vt[(16 + l15) * 72 + kk * 32 + half * 8];
      v16bf vb1 = cat8(*(const v8bf*)pv1, *(const v8bf*)(pv1 + 16));
      o0 = __builtin_amdgcn_wmma_f32_16x16x32_bf16(false, pa, false, vb0,
                                                   (short)0, o0, false, false);
      o1 = __builtin_amdgcn_wmma_f32_16x16x32_bf16(false, pa, false, vb1,
                                                   (short)0, o1, false, false);
    }
  }

  // ---- normalize and store O (bf16, [tok][h*32+d]) ----
#pragma unroll
  for (int r = 0; r < 8; ++r) {
    float inv = 1.0f / lrow[r];
    size_t tok = segTok + (size_t)qc * 128 + wave * 16 + half * 8 + r;
    O[tok * 384 + h * 32 + l15]      = (__bf16)(o0[r] * inv);
    O[tok * 384 + h * 32 + 16 + l15] = (__bf16)(o1[r] * inv);
  }
}

// ---------------------------------------------------------------------------
extern "C" void kernel_launch(void* const* d_in, const int* in_sizes, int n_in,
                              void* d_out, int out_size, void* d_ws, size_t ws_size,
                              hipStream_t stream)
{
  (void)in_sizes; (void)n_in; (void)out_size; (void)ws_size;
  const float* x  = (const float*)d_in[0];
  const float* Wr = (const float*)d_in[1];
  const float* br = (const float*)d_in[2];
  const float* Wq = (const float*)d_in[3];
  const float* bq = (const float*)d_in[4];
  const float* Wp = (const float*)d_in[5];
  const float* bp = (const float*)d_in[6];
  float* out = (float*)d_out;

  char* ws = (char*)d_ws;
  __bf16* WrT  = (__bf16*)(ws + 0);         //  384 x 768  (589,824 B)
  __bf16* WqT  = (__bf16*)(ws + 589824);    // 1152 x 384  (884,736 B)
  __bf16* WpT  = (__bf16*)(ws + 1474560);   //  768 x 384  (589,824 B)
  __bf16* XR   = (__bf16*)(ws + 2064384);   // 16384 x 384  (12.6 MB)
  __bf16* QKV  = (__bf16*)(ws + 14647296);  // 16384 x 1152 (37.7 MB)
  __bf16* Obuf = (__bf16*)(ws + 52396032);  // 16384 x 384  (12.6 MB)
  // total workspace: 64,978,944 bytes

  wt_convert<<<(384 * 768 + 255) / 256, 256, 0, stream>>>(Wr, WrT, 768, 384);
  wt_convert<<<(1152 * 384 + 255) / 256, 256, 0, stream>>>(Wq, WqT, 384, 1152);
  wt_convert<<<(768 * 384 + 255) / 256, 256, 0, stream>>>(Wp, WpT, 384, 768);

  dim3 blk(256);
  // xr = x @ W_reduce + b_reduce           (f32 A -> bf16 out)
  gemm_bf16_wmma<true, false><<<dim3(3, 128), blk, 0, stream>>>(
      x, WrT, br, XR, 16384, 384, 768);
  // qkv = xr @ W_qkv + b_qkv               (bf16 A -> bf16 out)
  gemm_bf16_wmma<false, false><<<dim3(9, 128), blk, 0, stream>>>(
      XR, WqT, bq, QKV, 16384, 1152, 384);
  // blocked flash attention                (4*12*4 blocks x 8 q-chunks)
  attn_wmma<<<1536, blk, 0, stream>>>(QKV, Obuf);
  // out = o @ W_proj + b_proj              (bf16 A -> f32 out)
  gemm_bf16_wmma<false, true><<<dim3(6, 128), blk, 0, stream>>>(
      Obuf, WpT, bp, out, 16384, 768, 384);
}
